// XYZEncoder_58695023067147
// MI455X (gfx1250) — matrix operand order repaired and checked by reading, not compiled, not run on hardware
//
#include <hip/hip_runtime.h>
#include <hip/hip_bf16.h>

typedef _Float16 h8   __attribute__((ext_vector_type(8)));
typedef _Float16 v16h __attribute__((ext_vector_type(16)));
typedef float    v8f  __attribute__((ext_vector_type(8)));

#define LVLS   16
#define TSIZE  (1u << 19)
#define TMASK  (TSIZE - 1u)

__device__ __constant__ float kRes[LVLS] = {
    16.f, 21.f, 27.f, 36.f, 48.f, 64.f, 84.f, 111.f,
    147.f, 194.f, 256.f, 337.f, 445.f, 588.f, 776.f, 1024.f};

__global__ __launch_bounds__(256) void ngp_fused_kernel(
    const float* __restrict__ x,      // [N,3]
    const float* __restrict__ table,  // [16, 2^19, 2]
    const float* __restrict__ W1,     // [32,64]
    const float* __restrict__ W2,     // [64,16]
    float* __restrict__ out,          // sigmas[N] ++ h[N,16]
    int N)
{
    __shared__ _Float16 sEnc[256][32];   // encodings, f16, [point][l*2+f]
    __shared__ _Float16 sW1t[64][32];    // W1 transposed: [n][k]
    __shared__ _Float16 sW2t[16][64];    // W2 transposed: [n][k]
    __shared__ _Float16 sHid[256][64];   // relu(enc@W1), f16

    const int tid = threadIdx.x;

    // ---- stage weights to LDS (transposed, f32 -> f16) ----
    for (int i = tid; i < 32 * 64; i += 256) {
        int k = i >> 6, n = i & 63;
        sW1t[n][k] = (_Float16)W1[i];
    }
    for (int i = tid; i < 64 * 16; i += 256) {
        int k = i >> 4, n = i & 15;
        sW2t[n][k] = (_Float16)W2[i];
    }

    // ---- hash-grid encode: one point per thread ----
    int p = blockIdx.x * 256 + tid;
    if (p >= N) p = N - 1;                 // keep EXEC full for WMMA phase
    const float px = x[3 * p + 0];
    const float py = x[3 * p + 1];
    const float pz = x[3 * p + 2];

    for (int l = 0; l < LVLS; ++l) {
        const float res = kRes[l];
        const float fx = px * res, fy = py * res, fz = pz * res;
        const float x0f = floorf(fx), y0f = floorf(fy), z0f = floorf(fz);
        const float wx = fx - x0f, wy = fy - y0f, wz = fz - z0f;
        const unsigned cx = (unsigned)x0f, cy = (unsigned)y0f, cz = (unsigned)z0f;
        const unsigned hx0 = cx,                       hx1 = cx + 1u;
        const unsigned hy0 = cy * 2654435761u,         hy1 = (cy + 1u) * 2654435761u;
        const unsigned hz0 = cz * 805459861u,          hz1 = (cz + 1u) * 805459861u;
        const float* tl = table + (size_t)l * TSIZE * 2;
        float f0 = 0.f, f1 = 0.f;
#pragma unroll
        for (int c = 0; c < 8; ++c) {
            const unsigned hx = (c & 4) ? hx1 : hx0;
            const unsigned hy = (c & 2) ? hy1 : hy0;
            const unsigned hz = (c & 1) ? hz1 : hz0;
            const unsigned idx = (hx ^ hy ^ hz) & TMASK;
            const float2 f = *(const float2*)(tl + 2 * (size_t)idx);
            const float w = ((c & 4) ? wx : 1.f - wx) *
                            ((c & 2) ? wy : 1.f - wy) *
                            ((c & 1) ? wz : 1.f - wz);
            f0 += w * f.x;
            f1 += w * f.y;
        }
        sEnc[tid][2 * l + 0] = (_Float16)f0;
        sEnc[tid][2 * l + 1] = (_Float16)f1;
    }

    __syncthreads();

    // ---- fused MLP via WMMA: each wave owns its 32 points ----
    const int wv    = tid >> 5;
    const int lane  = tid & 31;
    const int col   = lane & 15;               // B/C column, A row
    const int kbA   = (lane < 16) ? 0 : 8;     // A K-base (K in {kbA..kbA+7, kbA+16..kbA+23})
    const int kbB   = (lane < 16) ? 0 : 8;     // B K-base (K in {kbB..kbB+7, kbB+16..kbB+23})
    const int rbase = (lane < 16) ? 0 : 8;     // C/D row base

#pragma unroll
    for (int t = 0; t < 2; ++t) {
        const int rowA = wv * 32 + t * 16 + col;   // LDS row of this lane's A row

        // A = enc tile 16x32 (f16)
        h8 alo = *(const h8*)&sEnc[rowA][kbA];
        h8 ahi = *(const h8*)&sEnc[rowA][kbA + 16];
        v16h a = __builtin_shufflevector(alo, ahi,
                 0,1,2,3,4,5,6,7,8,9,10,11,12,13,14,15);

        // Layer 1: 4 WMMAs cover the 64 output columns
#pragma unroll
        for (int nb = 0; nb < 4; ++nb) {
            h8 b0 = *(const h8*)&sW1t[nb * 16 + col][kbB];
            h8 b1 = *(const h8*)&sW1t[nb * 16 + col][kbB + 16];
            v16h b = __builtin_shufflevector(b0, b1,
                     0,1,2,3,4,5,6,7,8,9,10,11,12,13,14,15);
            v8f czero = {};
            v8f c = __builtin_amdgcn_wmma_f32_16x16x32_f16(
                        false, a, false, b, (short)0, czero, false, false);
#pragma unroll
            for (int r = 0; r < 8; ++r) {
                float v = c[r];
                v = v > 0.f ? v : 0.f;   // ReLU
                sHid[wv * 32 + t * 16 + rbase + r][nb * 16 + col] = (_Float16)v;
            }
        }

        // Layer 2: K=64 -> two chained 16x16x32 WMMAs
        h8 a0 = *(const h8*)&sHid[rowA][kbA];
        h8 a1 = *(const h8*)&sHid[rowA][kbA + 16];
        h8 a2 = *(const h8*)&sHid[rowA][kbA + 32];
        h8 a3 = *(const h8*)&sHid[rowA][kbA + 48];
        v16h aL = __builtin_shufflevector(a0, a1,
                  0,1,2,3,4,5,6,7,8,9,10,11,12,13,14,15);
        v16h aH = __builtin_shufflevector(a2, a3,
                  0,1,2,3,4,5,6,7,8,9,10,11,12,13,14,15);

        h8 bl0 = *(const h8*)&sW2t[col][kbB];
        h8 bl1 = *(const h8*)&sW2t[col][kbB + 16];
        h8 bh0 = *(const h8*)&sW2t[col][32 + kbB];
        h8 bh1 = *(const h8*)&sW2t[col][32 + kbB + 16];
        v16h bL = __builtin_shufflevector(bl0, bl1,
                  0,1,2,3,4,5,6,7,8,9,10,11,12,13,14,15);
        v16h bH = __builtin_shufflevector(bh0, bh1,
                  0,1,2,3,4,5,6,7,8,9,10,11,12,13,14,15);

        v8f hzero = {};
        v8f hacc = __builtin_amdgcn_wmma_f32_16x16x32_f16(
                       false, aL, false, bL, (short)0, hzero, false, false);
        hacc = __builtin_amdgcn_wmma_f32_16x16x32_f16(
                       false, aH, false, bH, (short)0, hacc, false, false);

        // ---- write h (coalesced 64B runs) and sigma = exp(h[:,0]) ----
        const int prow0 = blockIdx.x * 256 + wv * 32 + t * 16 + rbase;
#pragma unroll
        for (int r = 0; r < 8; ++r) {
            const int prow = prow0 + r;
            if (prow < N) {
                const float v = hacc[r];
                out[(size_t)N + (size_t)prow * 16 + col] = v;
                if (col == 0) out[prow] = expf(v);
            }
        }
    }
}

extern "C" void kernel_launch(void* const* d_in, const int* in_sizes, int n_in,
                              void* d_out, int out_size, void* d_ws, size_t ws_size,
                              hipStream_t stream) {
    const float* x     = (const float*)d_in[0];
    const float* table = (const float*)d_in[1];
    const float* W1    = (const float*)d_in[2];
    const float* W2    = (const float*)d_in[3];
    float* out = (float*)d_out;
    const int N = in_sizes[0] / 3;                // x is [N,3]
    const int blocks = (N + 255) / 256;
    ngp_fused_kernel<<<blocks, 256, 0, stream>>>(x, table, W1, W2, out, N);
}
